// AutoformerBase_87703232184655
// MI455X (gfx1250) — compile-verified, matrix operand order checked by reading, not run on hardware
//
#include <hip/hip_runtime.h>
#include <hip/hip_bf16.h>

// ---------------------------------------------------------------------------
// Autoformer auto-correlation block for gfx1250 (MI455X), wave32 + WMMA.
// B=32, L=1024, D=512, H=8, E=64, top_k=6.
//
// mean over (h,e) of the circular correlation == diagonal sums of the batched
// Gram matrix Q_b K_b^T  ->  pure WMMA GEMM work, no FFT needed.
// ---------------------------------------------------------------------------

typedef __attribute__((ext_vector_type(16))) _Float16 v16h;
typedef __attribute__((ext_vector_type(8)))  _Float16 v8h;
typedef __attribute__((ext_vector_type(8)))  float    v8f;

#define BATCH 32
#define SEQL  1024
#define DIM   512
#define TOPK  6

__device__ __forceinline__ v16h cat16(v8h lo, v8h hi) {
  v16h r;
#pragma unroll
  for (int i = 0; i < 8; ++i) { r[i] = lo[i]; r[i + 8] = hi[i]; }
  return r;
}

// ---------------------------------------------------------------------------
// Generic M x 512 x 512 GEMM:  Out = A * W + bias
//   A_F16  : A operand is _Float16 (else float, converted on LDS stage)
//   OUT_F32: Out is float (else _Float16)
// Workgroup: 128 threads = 4 waves, 64(M) x 64(N) tile, K-step 32.
// Each wave owns a 16-wide N slice, 4 row-blocks -> 4 v8f accumulators,
// 4 x v_wmma_f32_16x16x32_f16 per K-step (B fragment reused across blocks).
// ---------------------------------------------------------------------------
template <bool A_F16, bool OUT_F32>
__global__ __launch_bounds__(128)
void gemm_wmma_k(const void* __restrict__ Aptr,
                 const float* __restrict__ W,
                 const float* __restrict__ bias,
                 void* __restrict__ Out) {
  alignas(16) __shared__ _Float16 Ah[64 * 32];   // [row][k]
  alignas(16) __shared__ _Float16 Bt[64 * 32];   // [col][k]  (transposed)

  const int tid  = threadIdx.x;
  const int wv   = tid >> 5;
  const int lane = tid & 31;
  const int m0   = (blockIdx.x >> 3) << 6;   // 512 M-tiles
  const int n0   = (blockIdx.x & 7) << 6;    // 8 N-tiles

  const int mm   = lane & 15;
  const int hi   = lane >> 4;
  const int base = hi ? 8 : 0;
  const int col  = n0 + (wv << 4) + mm;      // output column of this lane

  const float bval = bias[col];
  v8f acc[4];
#pragma unroll
  for (int rb = 0; rb < 4; ++rb)
#pragma unroll
    for (int e = 0; e < 8; ++e) acc[rb][e] = bval;

  for (int kk = 0; kk < DIM; kk += 32) {
    // Stage A tile 64x32 (convert f32->f16 if needed)
#pragma unroll
    for (int e = 0; e < 16; ++e) {
      const int idx = e * 128 + tid;
      const int r = idx >> 5, c = idx & 31;
      float av;
      if (A_F16)
        av = (float)((const _Float16*)Aptr)[(size_t)(m0 + r) * DIM + kk + c];
      else
        av = ((const float*)Aptr)[(size_t)(m0 + r) * DIM + kk + c];
      Ah[r * 32 + c] = (_Float16)av;
    }
    // Stage W tile 32x64, transposed so fragments are contiguous
#pragma unroll
    for (int e = 0; e < 16; ++e) {
      const int idx = e * 128 + tid;
      const int r = idx >> 6, c = idx & 63;
      Bt[c * 32 + r] = (_Float16)W[(size_t)(kk + r) * DIM + n0 + c];
    }
    __syncthreads();

    // B fragment (shared by all 4 row-blocks of this wave)
    const _Float16* bp = &Bt[((wv << 4) + mm) * 32 + base];
    const v16h bf = cat16(*(const v8h*)bp, *(const v8h*)(bp + 16));

#pragma unroll
    for (int rb = 0; rb < 4; ++rb) {
      const _Float16* ap = &Ah[(rb * 16 + mm) * 32 + base];
      const v16h af = cat16(*(const v8h*)ap, *(const v8h*)(ap + 16));
      acc[rb] = __builtin_amdgcn_wmma_f32_16x16x32_f16(
          false, af, false, bf, (short)0, acc[rb], false, false);
    }
    __syncthreads();
  }

  // Epilogue: row = m0 + rb*16 + r + 8*hi, col fixed per lane
#pragma unroll
  for (int rb = 0; rb < 4; ++rb)
#pragma unroll
    for (int r = 0; r < 8; ++r) {
      const int row = m0 + rb * 16 + r + 8 * hi;
      if (OUT_F32)
        ((float*)Out)[(size_t)row * DIM + col] = acc[rb][r];
      else
        ((_Float16*)Out)[(size_t)row * DIM + col] = (_Float16)acc[rb][r];
    }
}

// ---------------------------------------------------------------------------
// Fused Gram-diagonal correlation:
//   mean_value[b,l] += (1/512) * sum_{t-s = l (mod L)} Q[b,t,:].K[b,s,:]
// One workgroup (4 waves) per (b, 16-row t-strip). Each wave sweeps k-tiles,
// 16 WMMAs per tile over D=512, scatters the 16x16 tile into LDS lag bins.
// ---------------------------------------------------------------------------
__global__ __launch_bounds__(128)
void corr_k(const _Float16* __restrict__ Qh,
            const _Float16* __restrict__ Kh,
            float* __restrict__ mv) {
  __shared__ float lag[SEQL];
  const int tid  = threadIdx.x;
  const int wv   = tid >> 5;
  const int lane = tid & 31;
  const int b    = blockIdx.x >> 6;
  const int t0   = (blockIdx.x & 63) << 4;

  for (int i = tid; i < SEQL; i += 128) lag[i] = 0.0f;
  __syncthreads();

  const int mm   = lane & 15;
  const int hi   = lane >> 4;
  const int base = hi ? 8 : 0;

  const _Float16* qrow = Qh + (size_t)((b << 10) + t0 + mm) * DIM;

  for (int kt = wv; kt < 64; kt += 4) {           // uniform per wave
    const int k0 = kt << 4;
    const _Float16* krow = Kh + (size_t)((b << 10) + k0 + mm) * DIM;
    v8f acc = {};
#pragma unroll
    for (int kk = 0; kk < DIM; kk += 32) {
      const v16h af = cat16(*(const v8h*)(qrow + kk + base),
                            *(const v8h*)(qrow + kk + 16 + base));
      const v16h bf = cat16(*(const v8h*)(krow + kk + base),
                            *(const v8h*)(krow + kk + 16 + base));
      acc = __builtin_amdgcn_wmma_f32_16x16x32_f16(
          false, af, false, bf, (short)0, acc, false, false);
    }
    // C tile entry (i=r+8*hi, j=lane&15) -> lag (t0-k0+i-j) mod 1024
#pragma unroll
    for (int r = 0; r < 8; ++r) {
      const int l = (t0 - k0 + r + 8 * hi - mm + SEQL) & (SEQL - 1);
      atomicAdd(&lag[l], acc[r]);                 // ds_add_f32
    }
  }
  __syncthreads();

  for (int i = tid; i < SEQL; i += 128)
    atomicAdd(&mv[(b << 10) + i], lag[i] * (1.0f / 512.0f));
}

// ---------------------------------------------------------------------------
// Batch-mean over b, iterative top-6 argmax (tie-break: lower index),
// then per-batch softmax over the 6 selected lags.
// ---------------------------------------------------------------------------
__global__ __launch_bounds__(1024)
void topk_softmax_k(const float* __restrict__ mv,
                    int* __restrict__ idx_out,
                    float* __restrict__ w_out) {
  __shared__ float cm[SEQL];
  __shared__ float sv[SEQL];
  __shared__ int   si[SEQL];
  __shared__ int   sel[8];

  const int t = threadIdx.x;
  float s = 0.0f;
  for (int b = 0; b < BATCH; ++b) s += mv[(b << 10) + t];
  cm[t] = s * (1.0f / 32.0f);
  __syncthreads();

  for (int k = 0; k < TOPK; ++k) {
    bool taken = false;
    for (int j = 0; j < k; ++j) if (sel[j] == t) taken = true;
    sv[t] = taken ? -3.4e38f : cm[t];
    si[t] = t;
    __syncthreads();
    for (int stp = 512; stp > 0; stp >>= 1) {
      if (t < stp) {
        const float v2 = sv[t + stp];
        const int   i2 = si[t + stp];
        if (v2 > sv[t] || (v2 == sv[t] && i2 < si[t])) { sv[t] = v2; si[t] = i2; }
      }
      __syncthreads();
    }
    if (t == 0) sel[k] = si[0];
    __syncthreads();
  }

  if (t < TOPK) idx_out[t] = sel[t];
  if (t < BATCH) {
    float w[TOPK];
    float mx = -3.4e38f;
#pragma unroll
    for (int j = 0; j < TOPK; ++j) { w[j] = mv[(t << 10) + sel[j]]; mx = fmaxf(mx, w[j]); }
    float ssum = 0.0f;
#pragma unroll
    for (int j = 0; j < TOPK; ++j) { w[j] = __expf(w[j] - mx); ssum += w[j]; }
#pragma unroll
    for (int j = 0; j < TOPK; ++j) w_out[t * TOPK + j] = w[j] / ssum;
  }
}

// ---------------------------------------------------------------------------
// mid[b,l,:] = sum_j w[b,j] * V[b,(l+idx_j)%L,:]   (8 halves per thread)
// ---------------------------------------------------------------------------
__global__ __launch_bounds__(256)
void gather_combine_k(const _Float16* __restrict__ Vh,
                      const int* __restrict__ idx6,
                      const float* __restrict__ w,
                      _Float16* __restrict__ mid) {
  const int gid = blockIdx.x * 256 + threadIdx.x;
  const int d8 = gid & 63;
  const int l  = (gid >> 6) & (SEQL - 1);
  const int b  = gid >> 16;

  float acc[8] = {};
#pragma unroll
  for (int j = 0; j < TOPK; ++j) {
    const int row = (l + idx6[j]) & (SEQL - 1);
    const v8h vv = *(const v8h*)(Vh + (size_t)((b << 10) + row) * DIM + d8 * 8);
    const float wj = w[b * TOPK + j];
#pragma unroll
    for (int e = 0; e < 8; ++e) acc[e] += wj * (float)vv[e];
  }
  v8h o;
#pragma unroll
  for (int e = 0; e < 8; ++e) o[e] = (_Float16)acc[e];
  *(v8h*)(mid + (size_t)((b << 10) + l) * DIM + d8 * 8) = o;
}

__global__ void zero_f32_k(float* p, int n) {
  const int i = blockIdx.x * 256 + threadIdx.x;
  if (i < n) p[i] = 0.0f;
}

// ---------------------------------------------------------------------------
extern "C" void kernel_launch(void* const* d_in, const int* in_sizes, int n_in,
                              void* d_out, int out_size, void* d_ws, size_t ws_size,
                              hipStream_t stream) {
  (void)in_sizes; (void)n_in; (void)out_size; (void)ws_size;
  const float* q_in = (const float*)d_in[0];
  const float* k_in = (const float*)d_in[1];
  const float* v_in = (const float*)d_in[2];
  const float* Wq   = (const float*)d_in[3];
  const float* bq   = (const float*)d_in[4];
  const float* Wk   = (const float*)d_in[5];
  const float* bk   = (const float*)d_in[6];
  const float* Wv   = (const float*)d_in[7];
  const float* bv   = (const float*)d_in[8];
  const float* Wo   = (const float*)d_in[9];
  const float* bo   = (const float*)d_in[10];

  // Workspace layout (all 256B-aligned):
  //   [0, 32MB)    Qh (f16) -- reused as `mid` after corr_k
  //   [32, 64MB)   Kh (f16)
  //   [64, 96MB)   Vh (f16)
  //   +0KB..128KB  mean_value (32x1024 f32)
  //   + idx6 (6 int), tmp_corr (32x6 f32)
  char* ws = (char*)d_ws;
  const size_t SZH = (size_t)BATCH * SEQL * DIM * sizeof(_Float16);  // 32MB
  _Float16* Qh  = (_Float16*)(ws);
  _Float16* Kh  = (_Float16*)(ws + SZH);
  _Float16* Vh  = (_Float16*)(ws + 2 * SZH);
  float*    mv  = (float*)(ws + 3 * SZH);
  int*      idx = (int*)(ws + 3 * SZH + 131072);
  float*    tcw = (float*)(ws + 3 * SZH + 131072 + 256);
  _Float16* mid = Qh;  // Q no longer needed after corr_k

  zero_f32_k<<<128, 256, 0, stream>>>(mv, BATCH * SEQL);

  // Projections: (B*L x D) @ (D x D) + bias -> f16
  gemm_wmma_k<false, false><<<4096, 128, 0, stream>>>(q_in, Wq, bq, Qh);
  gemm_wmma_k<false, false><<<4096, 128, 0, stream>>>(k_in, Wk, bk, Kh);
  gemm_wmma_k<false, false><<<4096, 128, 0, stream>>>(v_in, Wv, bv, Vh);

  // Fused Gram-diagonal correlation -> mean_value
  corr_k<<<BATCH * 64, 128, 0, stream>>>(Qh, Kh, mv);

  // Top-6 lags + per-batch softmax weights
  topk_softmax_k<<<1, 1024, 0, stream>>>(mv, idx, tcw);

  // Weighted gather of shifted V -> mid (f16, overwrites Q buffer)
  gather_combine_k<<<8192, 256, 0, stream>>>(Vh, idx, tcw, mid);

  // Output projection: mid @ Wo + bo -> f32 d_out
  gemm_wmma_k<true, true><<<4096, 128, 0, stream>>>(mid, Wo, bo, (float*)d_out);
}
